// Model_48524540510507
// MI455X (gfx1250) — matrix operand order repaired and checked by reading, not compiled, not run on hardware
//
#include <hip/hip_runtime.h>
#include <hip/hip_bf16.h>

// ---------------------------------------------------------------------------
// Problem constants
// ---------------------------------------------------------------------------
constexpr int TT  = 1024;   // timesteps
constexpr int BB  = 16;     // batch (== WMMA M tile!)
constexpr int CII = 512;    // cond input
constexpr int ARR = 256;    // prenet width
constexpr int HH  = 1024;   // hidden
constexpr int GG  = 4096;   // 4*H gates
constexpr int DOO = 80;     // output dim (= 5 * 16)
constexpr int KXF = 24;     // 768/32 fragments for x
constexpr int FHH = 32;     // 1024/32 fragments for h-sized K
constexpr float LN_EPS = 1e-5f;

typedef __bf16 v16bf __attribute__((ext_vector_type(16)));
typedef float  v8f   __attribute__((ext_vector_type(8)));

// ---------------------------------------------------------------------------
// Helpers
// ---------------------------------------------------------------------------
__device__ __forceinline__ __bf16 f2bf(float f) {
    unsigned u = __builtin_bit_cast(unsigned, f);
    unsigned lsb = (u >> 16) & 1u;
    u += 0x7FFFu + lsb;                       // round-to-nearest-even
    unsigned short h = (unsigned short)(u >> 16);
    return __builtin_bit_cast(__bf16, h);
}

// A-fragment element offset (16-bit A 16x32 layout, ISA 7.12.2):
//  lanes 0-15 : row M=lane,    regs -> K 0..7 then 16..23
//  lanes 16-31: row M=lane-16, regs -> K 8..15 then 24..31
__device__ __forceinline__ int apack_off(int m, int k) {
    int kf = k >> 5;
    int kk = k & 31;
    int h, e;
    if (kk < 16) { h = kk >> 3; e = kk & 7; }
    else         { int k2 = kk - 16; h = k2 >> 3; e = 8 + (k2 & 7); }
    int lane = (m & 15) + (h << 4);
    return kf * 512 + lane * 16 + e;          // bf16 element units
}

__device__ __forceinline__ float sigm(float x) {
    return 1.0f / (1.0f + __expf(-x));
}

__device__ __forceinline__ v8f wmma_bf16(v16bf a, v16bf b, v8f c) {
    return __builtin_amdgcn_wmma_f32_16x16x32_bf16(
        /*neg_a=*/false, a, /*neg_b=*/false, b,
        /*c_mod=*/(short)0, c, /*reuse_a=*/false, /*reuse_b=*/false);
}

// CDNA5 async global->LDS copy: one instruction moves 16B per lane (512B/wave),
// tracked by ASYNCcnt (ISA 10.x / 15.18.3 opcode 98).
__device__ __forceinline__ void async_copy_b128(unsigned lds_off, const void* gptr) {
    unsigned long long ga = (unsigned long long)(size_t)gptr;
    asm volatile("global_load_async_to_lds_b128 %0, %1, off"
                 :: "v"(lds_off), "v"(ga) : "memory");
}
__device__ __forceinline__ void wait_async0() {
    asm volatile("s_wait_asynccnt 0x0" ::: "memory");
}

// Device-wide barrier for the 16 persistent workgroups. Also issues the CDNA5
// cluster split-barrier (real HW path when launched as a 16-WG cluster; the
// ISA defines it as S_NOP when ClusterID==0, so it is always safe).
__device__ __forceinline__ void grid_barrier(unsigned* cnt, unsigned* gen) {
    __builtin_amdgcn_s_cluster_barrier();
    __syncthreads();
    if (threadIdx.x == 0) {
        __threadfence();
        unsigned g = __hip_atomic_load(gen, __ATOMIC_RELAXED, __HIP_MEMORY_SCOPE_AGENT);
        unsigned a = __hip_atomic_fetch_add(cnt, 1u, __ATOMIC_ACQ_REL, __HIP_MEMORY_SCOPE_AGENT);
        if (a == 15u) {
            __hip_atomic_store(cnt, 0u, __ATOMIC_RELAXED, __HIP_MEMORY_SCOPE_AGENT);
            __hip_atomic_fetch_add(gen, 1u, __ATOMIC_RELEASE, __HIP_MEMORY_SCOPE_AGENT);
        } else {
            while (__hip_atomic_load(gen, __ATOMIC_ACQUIRE, __HIP_MEMORY_SCOPE_AGENT) == g)
                __builtin_amdgcn_s_sleep(2);
        }
    }
    __syncthreads();
}

// ---------------------------------------------------------------------------
// Weight pack: f32 (N,K) row-major -> bf16 B-fragment tiles (1024B/fragment,
// lane slice 32B: lanes 0-15 col N=lane K 0..7 then 16..23, lanes 16-31
// col N=lane-16 K 8..15 then 24..31).
// ---------------------------------------------------------------------------
__global__ void pack_w_kernel(const float* __restrict__ W, __bf16* __restrict__ dst,
                              int N, int K) {
    size_t idx = (size_t)blockIdx.x * blockDim.x + threadIdx.x;
    if (idx >= (size_t)N * K) return;
    int n = (int)(idx / K), k = (int)(idx % K);
    int nt = n >> 4, kf = k >> 5, kk = k & 31;
    int lane = (((kk >> 3) & 1) << 4) | (n & 15);
    int e = ((kk >> 4) << 3) | (kk & 7);
    dst[((size_t)nt * (K >> 5) + kf) * 512 + lane * 16 + e] = f2bf(W[idx]);
}

__global__ void add_bias_kernel(const float* __restrict__ a, const float* __restrict__ b,
                                float* __restrict__ d, int n) {
    int i = blockIdx.x * blockDim.x + threadIdx.x;
    if (i < n) d[i] = a[i] + b[i];
}

// ---------------------------------------------------------------------------
// Prenet + x_all A-fragment pack. One workgroup per timestep t.
// ---------------------------------------------------------------------------
__global__ __launch_bounds__(256)
void prenet_xpack_kernel(const float* __restrict__ cond, const float* __restrict__ targets,
                         const float* __restrict__ tmean, const float* __restrict__ tscale,
                         const float* __restrict__ Wpre1, const float* __restrict__ bpre1,
                         const float* __restrict__ Wpre2, const float* __restrict__ bpre2,
                         __bf16* __restrict__ xpack) {
    const int t = blockIdx.x;
    const int tid = threadIdx.x;
    __shared__ float s_prev[BB * DOO];
    __shared__ float s_h[BB * ARR];
    __shared__ float s_ar[BB * ARR];

    for (int j = tid; j < BB * DOO; j += 256) {
        int b = j / DOO, d = j % DOO;
        s_prev[j] = (t == 0) ? 0.0f
                  : (targets[((size_t)b * TT + (t - 1)) * DOO + d] - tmean[d]) / tscale[d];
    }
    __syncthreads();
    for (int j = tid; j < BB * ARR; j += 256) {
        int b = j >> 8, o = j & 255;
        float s = bpre1[o];
        for (int d = 0; d < DOO; ++d) s += s_prev[b * DOO + d] * Wpre1[o * DOO + d];
        s_h[j] = fmaxf(s, 0.0f);
    }
    __syncthreads();
    for (int j = tid; j < BB * ARR; j += 256) {
        int b = j >> 8, o = j & 255;
        float s = bpre2[o];
        for (int d = 0; d < ARR; ++d) s += s_h[b * ARR + d] * Wpre2[o * ARR + d];
        s_ar[j] = fmaxf(s, 0.0f);
    }
    __syncthreads();

    __bf16* xp = xpack + (size_t)t * (KXF * 512);
    for (int e0 = tid; e0 < KXF * 32; e0 += 256) {         // (frag, lane) pairs
        int kf = e0 >> 5, l = e0 & 31;
        int m = l & 15, half = l >> 4;
        for (int e = 0; e < 16; ++e) {
            int kk = (e < 8) ? (half * 8 + e) : (16 + half * 8 + (e - 8));
            int k = kf * 32 + kk;
            float v = (k < CII) ? cond[((size_t)m * TT + t) * CII + k]
                                : s_ar[m * ARR + (k - CII)];
            xp[(size_t)kf * 512 + l * 16 + e] = f2bf(v);
        }
    }
}

// ---------------------------------------------------------------------------
// LSTM cell + LayerNorm phase: workgroup b owns batch row b.
// ---------------------------------------------------------------------------
__device__ void lstm_ln_phase(const float* __restrict__ gates, float* __restrict__ c,
                              const float* __restrict__ lng, const float* __restrict__ lnb,
                              __bf16* __restrict__ lnpack, float* sh, float* red) {
    const int b = blockIdx.x;
    const int tid = threadIdx.x;
    const float* gr = gates + (size_t)b * GG;
    float* cr = c + (size_t)b * HH;

    for (int j = tid; j < HH; j += 256) {
        float iv = sigm(gr[j]);
        float fv = sigm(gr[HH + j]);
        float gv = tanhf(gr[2 * HH + j]);
        float ov = sigm(gr[3 * HH + j]);
        float c2 = fv * cr[j] + iv * gv;
        cr[j] = c2;
        sh[j] = ov * tanhf(c2);
    }
    __syncthreads();

    float s = 0.0f, s2 = 0.0f;
    for (int j = tid; j < HH; j += 256) { float v = sh[j]; s += v; s2 += v * v; }
    red[tid] = s; __syncthreads();
    for (int o = 128; o > 0; o >>= 1) { if (tid < o) red[tid] += red[tid + o]; __syncthreads(); }
    float mu = red[0] * (1.0f / HH);
    __syncthreads();
    red[tid] = s2; __syncthreads();
    for (int o = 128; o > 0; o >>= 1) { if (tid < o) red[tid] += red[tid + o]; __syncthreads(); }
    float var = red[0] * (1.0f / HH) - mu * mu;
    float rstd = rsqrtf(var + LN_EPS);
    __syncthreads();

    for (int j = tid; j < HH; j += 256) {
        float y = (sh[j] - mu) * rstd * lng[j] + lnb[j];
        lnpack[apack_off(b, j)] = f2bf(y);
    }
}

// 16x16 projection tile from LDS-staged A: D = A(16x1024)*Wtile + bias -> tanh.
__device__ void proj_tanh_pack(const __bf16* __restrict__ sA,
                               const __bf16* __restrict__ wtile,
                               const float* __restrict__ bias,
                               __bf16* __restrict__ dpack, int nt, int lane) {
    const int ln15 = lane & 15, lh = lane >> 4;
    const int nb = nt * 16;
    float bv = bias[nb + ln15];
    v8f acc; for (int r = 0; r < 8; ++r) acc[r] = bv;
    for (int kf = 0; kf < FHH; ++kf) {
        v16bf a = *(const v16bf*)(sA + (size_t)kf * 512 + lane * 16);
        v16bf b = *(const v16bf*)(wtile + (size_t)kf * 512 + lane * 16);
        __builtin_prefetch(wtile + (size_t)(kf + 2) * 512 + lane * 16, 0, 3);
        acc = wmma_bf16(a, b, acc);
    }
    for (int r = 0; r < 8; ++r) {
        int m = r + 8 * lh;                    // C/D layout: M = r + 8*(lane>=16)
        int k = nb + ln15;                     // output neuron index
        dpack[apack_off(m, k)] = f2bf(tanhf(acc[r]));
    }
}

// ---------------------------------------------------------------------------
// Persistent scan: 16 workgroups x 256 threads (128 wave32s), T steps.
// ---------------------------------------------------------------------------
__global__ __launch_bounds__(256)
void scan_kernel(const __bf16* __restrict__ xpack,
                 const __bf16* __restrict__ wih0p, const __bf16* __restrict__ whh0p,
                 const __bf16* __restrict__ wih1p, const __bf16* __restrict__ whh1p,
                 const __bf16* __restrict__ wp0p,  const __bf16* __restrict__ wp1p,
                 const __bf16* __restrict__ woutp,
                 const float* __restrict__ bias0, const float* __restrict__ bias1,
                 const float* __restrict__ bp0, const float* __restrict__ bp1,
                 const float* __restrict__ bout,
                 const float* __restrict__ lng0, const float* __restrict__ lnb0,
                 const float* __restrict__ lng1, const float* __restrict__ lnb1,
                 float* __restrict__ gates0, float* __restrict__ gates1,
                 float* __restrict__ ca, float* __restrict__ cb,
                 __bf16* __restrict__ za_pack, __bf16* __restrict__ zb_pack,
                 __bf16* __restrict__ ln0_pack, __bf16* __restrict__ ln1_pack,
                 unsigned* cnt, unsigned* gen, float* __restrict__ out) {
    const int tid  = threadIdx.x;
    const int lane = tid & 31;
    const int wave = tid >> 5;
    const int gw   = (int)blockIdx.x * 8 + wave;   // 0..127 global wave id
    const int ln15 = lane & 15;
    const int lh   = lane >> 4;

    __builtin_amdgcn_s_wait_tensorcnt(0);          // drain any TDM traffic

    // 64KB LDS: A-fragment staging buffer, unioned with LN scratch (phases
    // are barrier-separated so reuse is safe).
    __shared__ alignas(128) unsigned char s_mem[65536];
    __bf16* sA  = (__bf16*)s_mem;
    float*  sh  = (float*)s_mem;                   // 1024 floats
    float*  red = (float*)(s_mem + 4096);          // 256 floats
    const unsigned s_base = (unsigned)(size_t)&s_mem[0];   // LDS byte address

    for (int t = 0; t < TT; ++t) {
        // ---- P1: gates0 = x_t*Wih0^T + za*Whh0^T + b0 (256 n-tiles, 2/wave)
        {
            const __bf16* xp = xpack + (size_t)t * (KXF * 512);
            // async-stage shared A operands: x (24KB) then za (32KB)
            for (int o = tid * 16; o < KXF * 1024; o += 4096)
                async_copy_b128(s_base + o, (const char*)xp + o);
            for (int o = tid * 16; o < FHH * 1024; o += 4096)
                async_copy_b128(s_base + KXF * 1024 + o, (const char*)za_pack + o);
            wait_async0();
            __syncthreads();

            const int nt0 = gw * 2, nt1 = nt0 + 1;
            float bv0 = bias0[nt0 * 16 + ln15];
            float bv1 = bias0[nt1 * 16 + ln15];
            v8f acc0, acc1;
            for (int r = 0; r < 8; ++r) { acc0[r] = bv0; acc1[r] = bv1; }
            const __bf16* w0 = wih0p + (size_t)nt0 * (KXF * 512);
            const __bf16* w1 = wih0p + (size_t)nt1 * (KXF * 512);
            for (int kf = 0; kf < KXF; ++kf) {
                v16bf a  = *(const v16bf*)(sA + (size_t)kf * 512 + lane * 16);
                v16bf b0 = *(const v16bf*)(w0 + (size_t)kf * 512 + lane * 16);
                v16bf b1 = *(const v16bf*)(w1 + (size_t)kf * 512 + lane * 16);
                __builtin_prefetch(w0 + (size_t)(kf + 2) * 512 + lane * 16, 0, 3);
                __builtin_prefetch(w1 + (size_t)(kf + 2) * 512 + lane * 16, 0, 3);
                acc0 = wmma_bf16(a, b0, acc0);
                acc1 = wmma_bf16(a, b1, acc1);
            }
            const __bf16* z0 = whh0p + (size_t)nt0 * (FHH * 512);
            const __bf16* z1 = whh0p + (size_t)nt1 * (FHH * 512);
            for (int kf = 0; kf < FHH; ++kf) {
                v16bf a  = *(const v16bf*)(sA + (size_t)(KXF + kf) * 512 + lane * 16);
                v16bf b0 = *(const v16bf*)(z0 + (size_t)kf * 512 + lane * 16);
                v16bf b1 = *(const v16bf*)(z1 + (size_t)kf * 512 + lane * 16);
                __builtin_prefetch(z0 + (size_t)(kf + 2) * 512 + lane * 16, 0, 3);
                __builtin_prefetch(z1 + (size_t)(kf + 2) * 512 + lane * 16, 0, 3);
                acc0 = wmma_bf16(a, b0, acc0);
                acc1 = wmma_bf16(a, b1, acc1);
            }
            const int n0 = nt0 * 16 + ln15, n1 = nt1 * 16 + ln15;
            for (int r = 0; r < 8; ++r) {
                gates0[(size_t)(r + 8 * lh) * GG + n0] = acc0[r];
                gates0[(size_t)(r + 8 * lh) * GG + n1] = acc1[r];
            }
        }
        grid_barrier(cnt, gen);

        // ---- P2: LSTM0 + LN + pack ln0 (WG b owns batch row b)
        lstm_ln_phase(gates0, ca, lng0, lnb0, ln0_pack, sh, red);
        grid_barrier(cnt, gen);

        // ---- P3: za = tanh(ln0 * Wp0^T + bp0) (64 tiles, blocks 0..7)
        if (blockIdx.x < 8) {
            for (int o = tid * 16; o < FHH * 1024; o += 4096)
                async_copy_b128(s_base + o, (const char*)ln0_pack + o);
            wait_async0();
            __syncthreads();
            proj_tanh_pack(sA, wp0p + (size_t)gw * (FHH * 512), bp0, za_pack, gw, lane);
        }
        grid_barrier(cnt, gen);

        // ---- P4: gates1 = za*Wih1^T + zb*Whh1^T + b1
        {
            for (int o = tid * 16; o < FHH * 1024; o += 4096)
                async_copy_b128(s_base + o, (const char*)za_pack + o);
            for (int o = tid * 16; o < FHH * 1024; o += 4096)
                async_copy_b128(s_base + FHH * 1024 + o, (const char*)zb_pack + o);
            wait_async0();
            __syncthreads();

            const int nt0 = gw * 2, nt1 = nt0 + 1;
            float bv0 = bias1[nt0 * 16 + ln15];
            float bv1 = bias1[nt1 * 16 + ln15];
            v8f acc0, acc1;
            for (int r = 0; r < 8; ++r) { acc0[r] = bv0; acc1[r] = bv1; }
            const __bf16* w0 = wih1p + (size_t)nt0 * (FHH * 512);
            const __bf16* w1 = wih1p + (size_t)nt1 * (FHH * 512);
            for (int kf = 0; kf < FHH; ++kf) {
                v16bf a  = *(const v16bf*)(sA + (size_t)kf * 512 + lane * 16);
                v16bf b0 = *(const v16bf*)(w0 + (size_t)kf * 512 + lane * 16);
                v16bf b1 = *(const v16bf*)(w1 + (size_t)kf * 512 + lane * 16);
                __builtin_prefetch(w0 + (size_t)(kf + 2) * 512 + lane * 16, 0, 3);
                __builtin_prefetch(w1 + (size_t)(kf + 2) * 512 + lane * 16, 0, 3);
                acc0 = wmma_bf16(a, b0, acc0);
                acc1 = wmma_bf16(a, b1, acc1);
            }
            const __bf16* z0 = whh1p + (size_t)nt0 * (FHH * 512);
            const __bf16* z1 = whh1p + (size_t)nt1 * (FHH * 512);
            for (int kf = 0; kf < FHH; ++kf) {
                v16bf a  = *(const v16bf*)(sA + (size_t)(FHH + kf) * 512 + lane * 16);
                v16bf b0 = *(const v16bf*)(z0 + (size_t)kf * 512 + lane * 16);
                v16bf b1 = *(const v16bf*)(z1 + (size_t)kf * 512 + lane * 16);
                __builtin_prefetch(z0 + (size_t)(kf + 2) * 512 + lane * 16, 0, 3);
                __builtin_prefetch(z1 + (size_t)(kf + 2) * 512 + lane * 16, 0, 3);
                acc0 = wmma_bf16(a, b0, acc0);
                acc1 = wmma_bf16(a, b1, acc1);
            }
            const int n0 = nt0 * 16 + ln15, n1 = nt1 * 16 + ln15;
            for (int r = 0; r < 8; ++r) {
                gates1[(size_t)(r + 8 * lh) * GG + n0] = acc0[r];
                gates1[(size_t)(r + 8 * lh) * GG + n1] = acc1[r];
            }
        }
        grid_barrier(cnt, gen);

        // ---- P5: LSTM1 + LN + pack ln1
        lstm_ln_phase(gates1, cb, lng1, lnb1, ln1_pack, sh, red);
        grid_barrier(cnt, gen);

        // ---- P6: zb = tanh(ln1 * Wp1^T + bp1)
        if (blockIdx.x < 8) {
            for (int o = tid * 16; o < FHH * 1024; o += 4096)
                async_copy_b128(s_base + o, (const char*)ln1_pack + o);
            wait_async0();
            __syncthreads();
            proj_tanh_pack(sA, wp1p + (size_t)gw * (FHH * 512), bp1, zb_pack, gw, lane);
        }
        grid_barrier(cnt, gen);

        // ---- P7: out_t = zb * Wout^T + bout (5 n-tiles of 16 -> 80 dims)
        if (gw < 5) {
            const int nt = gw, nb = nt * 16;
            float bv = bout[nb + ln15];
            v8f acc; for (int r = 0; r < 8; ++r) acc[r] = bv;
            const __bf16* wt = woutp + (size_t)nt * (FHH * 512);
            for (int kf = 0; kf < FHH; ++kf) {
                v16bf a = *(const v16bf*)(zb_pack + (size_t)kf * 512 + lane * 16);
                v16bf b = *(const v16bf*)(wt + (size_t)kf * 512 + lane * 16);
                acc = wmma_bf16(a, b, acc);
            }
            const int n = nb + ln15;
            for (int r = 0; r < 8; ++r) {
                int m = r + 8 * lh;                       // batch row
                out[((size_t)m * TT + t) * DOO + n] = acc[r];
            }
        }
        // zb_pack written in P6 is barrier-protected before next-step P4 by
        // the in-step barriers; P7 runs concurrently with next P1 safely.
    }
}

// ---------------------------------------------------------------------------
// Host launch
// ---------------------------------------------------------------------------
extern "C" void kernel_launch(void* const* d_in, const int* in_sizes, int n_in,
                              void* d_out, int out_size, void* d_ws, size_t ws_size,
                              hipStream_t stream) {
    (void)in_sizes; (void)n_in; (void)out_size;
    const float* cond   = (const float*)d_in[0];
    const float* targets= (const float*)d_in[1];
    const float* tmean  = (const float*)d_in[2];
    const float* tscale = (const float*)d_in[3];
    const float* Wpre1  = (const float*)d_in[4];
    const float* bpre1  = (const float*)d_in[5];
    const float* Wpre2  = (const float*)d_in[6];
    const float* bpre2  = (const float*)d_in[7];
    const float* Wih0   = (const float*)d_in[8];
    const float* Whh0   = (const float*)d_in[9];
    const float* bih0   = (const float*)d_in[10];
    const float* bhh0   = (const float*)d_in[11];
    const float* lng0   = (const float*)d_in[12];
    const float* lnb0   = (const float*)d_in[13];
    const float* Wp0    = (const float*)d_in[14];
    const float* bp0    = (const float*)d_in[15];
    const float* Wih1   = (const float*)d_in[16];
    const float* Whh1   = (const float*)d_in[17];
    const float* bih1   = (const float*)d_in[18];
    const float* bhh1   = (const float*)d_in[19];
    const float* lng1   = (const float*)d_in[20];
    const float* lnb1   = (const float*)d_in[21];
    const float* Wp1    = (const float*)d_in[22];
    const float* bp1    = (const float*)d_in[23];
    const float* Wout   = (const float*)d_in[24];
    const float* bout   = (const float*)d_in[25];
    float* out = (float*)d_out;

    char* ws = (char*)d_ws;
    size_t off = 0;
    auto alloc = [&](size_t bytes) { size_t o = off; off += (bytes + 255) & ~(size_t)255; return o; };

    // state region (zeroed each call)
    size_t o_cnt = alloc(256);                       // [0]=arrive count, [64]=generation
    size_t o_ca  = alloc((size_t)BB * HH * 4);
    size_t o_cb  = alloc((size_t)BB * HH * 4);
    size_t o_zap = alloc((size_t)FHH * 512 * 2);
    size_t o_zbp = alloc((size_t)FHH * 512 * 2);
    size_t stateEnd = off;
    // scratch (written before read each step)
    size_t o_l0p = alloc((size_t)FHH * 512 * 2);
    size_t o_l1p = alloc((size_t)FHH * 512 * 2);
    size_t o_g0  = alloc((size_t)BB * GG * 4);
    size_t o_g1  = alloc((size_t)BB * GG * 4);
    size_t o_b0  = alloc((size_t)GG * 4);
    size_t o_b1  = alloc((size_t)GG * 4);
    // packed activations/weights
    size_t o_xp   = alloc((size_t)TT  * KXF * 512 * 2);
    size_t o_wih0 = alloc((size_t)256 * KXF * 512 * 2);
    size_t o_whh0 = alloc((size_t)256 * FHH * 512 * 2);
    size_t o_wih1 = alloc((size_t)256 * FHH * 512 * 2);
    size_t o_whh1 = alloc((size_t)256 * FHH * 512 * 2);
    size_t o_wp0  = alloc((size_t)64  * FHH * 512 * 2);
    size_t o_wp1  = alloc((size_t)64  * FHH * 512 * 2);
    size_t o_wout = alloc((size_t)5   * FHH * 512 * 2);
    if (ws_size < off) return;                       // insufficient workspace

    hipMemsetAsync(ws, 0, stateEnd, stream);

    auto bfp = [&](size_t o) { return (__bf16*)(ws + o); };
    auto fp  = [&](size_t o) { return (float*)(ws + o); };

    // pack weights to bf16 B-fragment tiles
    {
        auto launch_pack = [&](const float* W, size_t o, int N, int K) {
            size_t tot = (size_t)N * K;
            int blocks = (int)((tot + 255) / 256);
            pack_w_kernel<<<blocks, 256, 0, stream>>>(W, bfp(o), N, K);
        };
        launch_pack(Wih0, o_wih0, GG, CII + ARR);
        launch_pack(Whh0, o_whh0, GG, HH);
        launch_pack(Wih1, o_wih1, GG, HH);
        launch_pack(Whh1, o_whh1, GG, HH);
        launch_pack(Wp0,  o_wp0,  HH, HH);
        launch_pack(Wp1,  o_wp1,  HH, HH);
        launch_pack(Wout, o_wout, DOO, HH);
    }
    add_bias_kernel<<<GG / 256, 256, 0, stream>>>(bih0, bhh0, fp(o_b0), GG);
    add_bias_kernel<<<GG / 256, 256, 0, stream>>>(bih1, bhh1, fp(o_b1), GG);

    prenet_xpack_kernel<<<TT, 256, 0, stream>>>(cond, targets, tmean, tscale,
                                                Wpre1, bpre1, Wpre2, bpre2, bfp(o_xp));

    unsigned* cnt = (unsigned*)(ws + o_cnt);
    unsigned* gen = (unsigned*)(ws + o_cnt + 64);
    scan_kernel<<<16, 256, 0, stream>>>(bfp(o_xp),
                                        bfp(o_wih0), bfp(o_whh0), bfp(o_wih1), bfp(o_whh1),
                                        bfp(o_wp0), bfp(o_wp1), bfp(o_wout),
                                        fp(o_b0), fp(o_b1), bp0, bp1, bout,
                                        lng0, lnb0, lng1, lnb1,
                                        fp(o_g0), fp(o_g1), fp(o_ca), fp(o_cb),
                                        bfp(o_zap), bfp(o_zbp), bfp(o_l0p), bfp(o_l1p),
                                        cnt, gen, out);
}
